// MultiHeadAttention_6717328851522
// MI455X (gfx1250) — matrix operand order repaired and checked by reading, not compiled, not run on hardware
//
#include <hip/hip_runtime.h>

// MHA forward for MI455X (gfx1250, wave32, WMMA).
// Pipeline: [QKV gemm bf16-wmma] -> [flash attention bf16-wmma] -> [out gemm bf16-wmma]
// Matrix math on v_wmma_f32_16x16x32_bf16 (f32 accumulate).
// bf16 tile staging via GLOBAL_LOAD_ASYNC_TO_LDS_B128 (ASYNCcnt) where no
// transpose/convert is needed; manual ds stores elsewhere.

typedef __attribute__((ext_vector_type(16))) __bf16 v16bf;
typedef __attribute__((ext_vector_type(8)))  float  v8f;
typedef __attribute__((ext_vector_type(4)))  int    v4i_;

#define LOG2E 1.4426950408889634f
#define ATTN_SCALE 0.125f     // 64^-0.5

__device__ __forceinline__ unsigned short f2bfu(float f) {
  unsigned u = __builtin_bit_cast(unsigned, f);
  u += 0x7FFFu + ((u >> 16) & 1u);          // round-to-nearest-even
  return (unsigned short)(u >> 16);
}
__device__ __forceinline__ __bf16 u2bf(unsigned short s) {
  return __builtin_bit_cast(__bf16, s);
}

template <int MASK>
__device__ __forceinline__ float xswap(float x) {
  int i = __builtin_bit_cast(int, x);
  i = __builtin_amdgcn_ds_swizzle(i, (MASK << 10) | 0x1F);   // xor swizzle, group of 32
  return __builtin_bit_cast(float, i);
}

// 16-byte async global->LDS copy (gfx1250, tracked by ASYNCcnt).
__device__ __forceinline__ void async_copy16(const void* g, void* l) {
  __builtin_amdgcn_global_load_async_to_lds_b128(
      (__attribute__((address_space(1))) v4i_*)g,
      (__attribute__((address_space(3))) v4i_*)l, 0, 0);
}
__device__ __forceinline__ void wait_async0() {
  asm volatile("s_wait_asynccnt 0x0" ::: "memory");
}

// ---------------------------------------------------------------------------
// Kernel 1: QKV projection.  X[8192,1024] fp32 * W[1024,3072] fp32 + b
// -> Q/K/V bf16 laid out [B=4, H=16, S=2048, HD=64].
// Block tile 128x64, 256 threads (8 waves), wave tile 32x32 (2x2 WMMA).
// ---------------------------------------------------------------------------
#define G_BM 128
#define G_BN 64
#define G_BK 32
#define G_LDA 40     // ushort stride; 80B = multiple of 16 (b128-friendly)
#define G_LDB 36

__global__ __launch_bounds__(256) void qkv_gemm_kernel(
    const float* __restrict__ X, const float* __restrict__ W,
    const float* __restrict__ bias, unsigned short* __restrict__ qkv) {
  __shared__ unsigned short sA[G_BM * G_LDA];
  __shared__ unsigned short sB[G_BN * G_LDB];   // transposed: [n][k]

  const int tid  = threadIdx.x;
  const int wave = tid >> 5, lane = tid & 31;
  const int half = lane >> 4, mq = lane & 15;
  const int m0 = blockIdx.y * G_BM;
  const int n0 = blockIdx.x * G_BN;
  const int wm = wave & 3, wn = wave >> 2;

  v8f acc[2][2] = {};

  for (int k0 = 0; k0 < 1024; k0 += G_BK) {
    // Prefetch next k-tile while we convert/stage this one.
    if (k0 + G_BK < 1024) {
      __builtin_prefetch(X + (size_t)(m0 + (tid >> 1)) * 1024 + k0 + G_BK, 0, 0);
      __builtin_prefetch(W + (size_t)(k0 + G_BK + (tid >> 4)) * 3072 + n0, 0, 0);
    }
    __syncthreads();
    // Stage A tile (128x32 fp32 -> bf16), 1024 float4 units / 256 threads.
#pragma unroll
    for (int i = 0; i < 4; ++i) {
      int idx = tid + i * 256;
      int r = idx >> 3, c = (idx & 7) << 2;
      const float4 v = *(const float4*)(X + (size_t)(m0 + r) * 1024 + k0 + c);
      sA[r * G_LDA + c + 0] = f2bfu(v.x);
      sA[r * G_LDA + c + 1] = f2bfu(v.y);
      sA[r * G_LDA + c + 2] = f2bfu(v.z);
      sA[r * G_LDA + c + 3] = f2bfu(v.w);
    }
    // Stage B tile transposed (W[k0..+31][n0..+63] -> sB[n][k]).
#pragma unroll
    for (int i = 0; i < 2; ++i) {
      int idx = tid + i * 256;
      int r = idx >> 4, c = (idx & 15) << 2;
      const float4 v = *(const float4*)(W + (size_t)(k0 + r) * 3072 + n0 + c);
      sB[(c + 0) * G_LDB + r] = f2bfu(v.x);
      sB[(c + 1) * G_LDB + r] = f2bfu(v.y);
      sB[(c + 2) * G_LDB + r] = f2bfu(v.z);
      sB[(c + 3) * G_LDB + r] = f2bfu(v.w);
    }
    __syncthreads();

    // Gather fragments per ISA VGPR layouts (16-bit A 16x32, B 32x16).
    v16bf afrag[2], bfrag[2];
#pragma unroll
    for (int mi = 0; mi < 2; ++mi) {
      const unsigned short* ap = &sA[(wm * 32 + mi * 16 + mq) * G_LDA];
#pragma unroll
      for (int j = 0; j < 8; ++j) {
        int k = (j < 4) ? (half * 8 + 2 * j) : (16 + half * 8 + 2 * (j - 4));
        afrag[mi][2 * j]     = u2bf(ap[k]);
        afrag[mi][2 * j + 1] = u2bf(ap[k + 1]);
      }
    }
#pragma unroll
    for (int ni = 0; ni < 2; ++ni) {
      const unsigned short* bp = &sB[(wn * 32 + ni * 16 + mq) * G_LDB];
#pragma unroll
      for (int j = 0; j < 8; ++j) {
        int k = (j < 4) ? (half * 8 + 2 * j) : (16 + half * 8 + 2 * (j - 4));
        bfrag[ni][2 * j]     = u2bf(bp[k]);
        bfrag[ni][2 * j + 1] = u2bf(bp[k + 1]);
      }
    }
#pragma unroll
    for (int mi = 0; mi < 2; ++mi)
#pragma unroll
      for (int ni = 0; ni < 2; ++ni)
        acc[mi][ni] = __builtin_amdgcn_wmma_f32_16x16x32_bf16(
            false, afrag[mi], false, bfrag[ni], (short)0, acc[mi][ni], false, false);
  }

  // Epilogue: +bias, scatter bf16 to q/k/v [B,H,S,HD].
#pragma unroll
  for (int ni = 0; ni < 2; ++ni) {
    int n = n0 + wn * 32 + ni * 16 + mq;
    float bv = bias[n];
    int which = n >> 10, d = n & 1023;
    int h = d >> 6, hd = d & 63;
    unsigned short* dst = qkv + (size_t)which * (8192ull * 1024ull);
#pragma unroll
    for (int mi = 0; mi < 2; ++mi) {
#pragma unroll
      for (int r = 0; r < 8; ++r) {
        int row = m0 + wm * 32 + mi * 16 + r + half * 8;
        int b = row >> 11, s = row & 2047;
        dst[(((size_t)(b * 16 + h)) * 2048 + s) * 64 + hd] =
            f2bfu(acc[mi][ni][r] + bv);
      }
    }
  }
}

// ---------------------------------------------------------------------------
// Kernel 2: flash attention, one (b,h) per blockIdx.y, 64 query rows per block.
// 128 threads = 4 waves; each wave owns 16 query rows. KV tiles of 32.
// K tile staged with async global->LDS b128; V transposed manually.
// ---------------------------------------------------------------------------
#define F_KSTR 72    // ushort stride; 144B = multiple of 16 (b128-friendly)
#define F_VSTR 36
#define F_PSTR 34

__global__ __launch_bounds__(128) void flash_attn_kernel(
    const unsigned short* __restrict__ Q, const unsigned short* __restrict__ K,
    const unsigned short* __restrict__ V, unsigned short* __restrict__ O) {
  __shared__ unsigned short sK[32 * F_KSTR];       // [key][hd]
  __shared__ unsigned short sV[64 * F_VSTR];       // transposed: [hd][key]
  __shared__ unsigned short sP[4][16 * F_PSTR];    // per-wave P tile [row][key]

  const int tid  = threadIdx.x;
  const int wave = tid >> 5, lane = tid & 31;
  const int half = lane >> 4, mq = lane & 15;
  const int bh = blockIdx.y;                       // b*16 + h
  const int q0 = blockIdx.x * 64 + wave * 16;

  const unsigned short* qbase = Q + ((size_t)bh * 2048 + q0) * 64;
  const unsigned short* kbase = K + (size_t)bh * 2048 * 64;
  const unsigned short* vbase = V + (size_t)bh * 2048 * 64;

  // Q fragments: two 16x32 chunks of the 16x64 Q tile (A-matrix layout).
  v16bf aq[2];
#pragma unroll
  for (int c = 0; c < 2; ++c) {
    const unsigned short* qp = qbase + mq * 64 + c * 32;
#pragma unroll
    for (int j = 0; j < 8; ++j) {
      int k = (j < 4) ? (half * 8 + 2 * j) : (16 + half * 8 + 2 * (j - 4));
      aq[c][2 * j]     = u2bf(qp[k]);
      aq[c][2 * j + 1] = u2bf(qp[k + 1]);
    }
  }

  v8f acc[4] = {};
  float mrun[8], lrun[8];
#pragma unroll
  for (int r = 0; r < 8; ++r) { mrun[r] = -1e30f; lrun[r] = 0.f; }

  for (int kv0 = 0; kv0 < 2048; kv0 += 32) {
    __syncthreads();
    // K tile: 32x64 ushort = 256 x 16B chunks, async global->LDS.
#pragma unroll
    for (int i = 0; i < 2; ++i) {
      int idx = tid + i * 128;                     // 0..255
      int r = idx >> 3, c = (idx & 7) << 3;        // r = key, c = hd (8-ushort chunk)
      async_copy16(kbase + (size_t)(kv0 + r) * 64 + c, &sK[r * F_KSTR + c]);
    }
    // V tile: stage transposed [hd][key] (manual; transpose in flight).
#pragma unroll
    for (int i = 0; i < 4; ++i) {
      int idx = tid + i * 128;
      int r = idx >> 4, c = (idx & 15) << 2;       // r = key, c = hd
      const ushort4 vv = *(const ushort4*)(vbase + (size_t)(kv0 + r) * 64 + c);
      sV[(c + 0) * F_VSTR + r] = vv.x;
      sV[(c + 1) * F_VSTR + r] = vv.y;
      sV[(c + 2) * F_VSTR + r] = vv.z;
      sV[(c + 3) * F_VSTR + r] = vv.w;
    }
    wait_async0();
    __syncthreads();

    // S = Q * K^T for this pair of 16x16 score tiles.
    v8f sc[2] = {};
#pragma unroll
    for (int nt = 0; nt < 2; ++nt) {
#pragma unroll
      for (int c = 0; c < 2; ++c) {
        v16bf bk;
        const unsigned short* bp = &sK[(nt * 16 + mq) * F_KSTR + c * 32];
#pragma unroll
        for (int j = 0; j < 8; ++j) {
          int k = (j < 4) ? (half * 8 + 2 * j) : (16 + half * 8 + 2 * (j - 4));
          bk[2 * j]     = u2bf(bp[k]);
          bk[2 * j + 1] = u2bf(bp[k + 1]);
        }
        sc[nt] = __builtin_amdgcn_wmma_f32_16x16x32_bf16(
            false, aq[c], false, bk, (short)0, sc[nt], false, false);
      }
    }

    // Online softmax. C-layout: reg r holds row (r + 8*half), col = lane%16.
#pragma unroll
    for (int r = 0; r < 8; ++r) {
      float x0 = sc[0][r] * ATTN_SCALE;
      float x1 = sc[1][r] * ATTN_SCALE;
      float t = fmaxf(x0, x1);
      t = fmaxf(t, xswap<1>(t)); t = fmaxf(t, xswap<2>(t));
      t = fmaxf(t, xswap<4>(t)); t = fmaxf(t, xswap<8>(t));
      float mnew = fmaxf(mrun[r], t);
      float corr = exp2f((mrun[r] - mnew) * LOG2E);
      float p0 = exp2f((x0 - mnew) * LOG2E);
      float p1 = exp2f((x1 - mnew) * LOG2E);
      float rs = p0 + p1;
      rs += xswap<1>(rs); rs += xswap<2>(rs);
      rs += xswap<4>(rs); rs += xswap<8>(rs);
      lrun[r] = lrun[r] * corr + rs;
      mrun[r] = mnew;
#pragma unroll
      for (int nt2 = 0; nt2 < 4; ++nt2) acc[nt2][r] *= corr;
      int prow = r + half * 8;
      sP[wave][prow * F_PSTR + mq]      = f2bfu(p0);
      sP[wave][prow * F_PSTR + 16 + mq] = f2bfu(p1);
    }
    // Per-wave LDS RAW fence (DS ops are wave-ordered; make it explicit).
    asm volatile("s_wait_dscnt 0" ::: "memory");

    // Re-gather P as an A-matrix fragment (16 rows x 32 keys).
    v16bf pa;
    const unsigned short* pp = &sP[wave][mq * F_PSTR];
#pragma unroll
    for (int j = 0; j < 8; ++j) {
      int k = (j < 4) ? (half * 8 + 2 * j) : (16 + half * 8 + 2 * (j - 4));
      pa[2 * j]     = u2bf(pp[k]);
      pa[2 * j + 1] = u2bf(pp[k + 1]);
    }
    // O += P * V  (4 WMMAs across HD=64).
#pragma unroll
    for (int nt2 = 0; nt2 < 4; ++nt2) {
      v16bf bv;
      const unsigned short* vp = &sV[(nt2 * 16 + mq) * F_VSTR];
#pragma unroll
      for (int j = 0; j < 8; ++j) {
        int k = (j < 4) ? (half * 8 + 2 * j) : (16 + half * 8 + 2 * (j - 4));
        bv[2 * j]     = u2bf(vp[k]);
        bv[2 * j + 1] = u2bf(vp[k + 1]);
      }
      acc[nt2] = __builtin_amdgcn_wmma_f32_16x16x32_bf16(
          false, pa, false, bv, (short)0, acc[nt2], false, false);
    }
  }

  // Normalize and write O in [B,S,D] bf16 (head-concatenated for out proj).
  const int b = bh >> 4, h = bh & 15;
#pragma unroll
  for (int r = 0; r < 8; ++r) {
    float inv = 1.0f / lrun[r];
    int s_idx = q0 + r + half * 8;
    unsigned short* op = O + ((size_t)(b * 2048 + s_idx)) * 1024 + h * 64;
#pragma unroll
    for (int nt2 = 0; nt2 < 4; ++nt2)
      op[nt2 * 16 + mq] = f2bfu(acc[nt2][r] * inv);
  }
}

// ---------------------------------------------------------------------------
// Kernel 3: output projection.  O[8192,1024] bf16 * W[1024,1024] fp32 + b -> fp32.
// A tile (already bf16) staged with async global->LDS b128.
// ---------------------------------------------------------------------------
__global__ __launch_bounds__(256) void out_gemm_kernel(
    const unsigned short* __restrict__ A, const float* __restrict__ W,
    const float* __restrict__ bias, float* __restrict__ Out) {
  __shared__ unsigned short sA[G_BM * G_LDA];
  __shared__ unsigned short sB[G_BN * G_LDB];

  const int tid  = threadIdx.x;
  const int wave = tid >> 5, lane = tid & 31;
  const int half = lane >> 4, mq = lane & 15;
  const int m0 = blockIdx.y * G_BM;
  const int n0 = blockIdx.x * G_BN;
  const int wm = wave & 3, wn = wave >> 2;

  v8f acc[2][2] = {};

  for (int k0 = 0; k0 < 1024; k0 += G_BK) {
    __syncthreads();
    // A tile: 128x32 ushort = 512 x 16B chunks, async global->LDS.
#pragma unroll
    for (int i = 0; i < 2; ++i) {
      int idx = tid + i * 256;                     // 0..511
      int r = idx >> 2, c = (idx & 3) << 3;        // r = row, c = 8-ushort chunk
      async_copy16(A + (size_t)(m0 + r) * 1024 + k0 + c, &sA[r * G_LDA + c]);
    }
    // B tile transposed (fp32 -> bf16 in flight).
#pragma unroll
    for (int i = 0; i < 2; ++i) {
      int idx = tid + i * 256;
      int r = idx >> 4, c = (idx & 15) << 2;
      const float4 v = *(const float4*)(W + (size_t)(k0 + r) * 1024 + n0 + c);
      sB[(c + 0) * G_LDB + r] = f2bfu(v.x);
      sB[(c + 1) * G_LDB + r] = f2bfu(v.y);
      sB[(c + 2) * G_LDB + r] = f2bfu(v.z);
      sB[(c + 3) * G_LDB + r] = f2bfu(v.w);
    }
    wait_async0();
    __syncthreads();

    v16bf afrag[2], bfrag[2];
#pragma unroll
    for (int mi = 0; mi < 2; ++mi) {
      const unsigned short* ap = &sA[(wm * 32 + mi * 16 + mq) * G_LDA];
#pragma unroll
      for (int j = 0; j < 8; ++j) {
        int k = (j < 4) ? (half * 8 + 2 * j) : (16 + half * 8 + 2 * (j - 4));
        afrag[mi][2 * j]     = u2bf(ap[k]);
        afrag[mi][2 * j + 1] = u2bf(ap[k + 1]);
      }
    }
#pragma unroll
    for (int ni = 0; ni < 2; ++ni) {
      const unsigned short* bp = &sB[(wn * 32 + ni * 16 + mq) * G_LDB];
#pragma unroll
      for (int j = 0; j < 8; ++j) {
        int k = (j < 4) ? (half * 8 + 2 * j) : (16 + half * 8 + 2 * (j - 4));
        bfrag[ni][2 * j]     = u2bf(bp[k]);
        bfrag[ni][2 * j + 1] = u2bf(bp[k + 1]);
      }
    }
#pragma unroll
    for (int mi = 0; mi < 2; ++mi)
#pragma unroll
      for (int ni = 0; ni < 2; ++ni)
        acc[mi][ni] = __builtin_amdgcn_wmma_f32_16x16x32_bf16(
            false, afrag[mi], false, bfrag[ni], (short)0, acc[mi][ni], false, false);
  }

#pragma unroll
  for (int ni = 0; ni < 2; ++ni) {
    int n = n0 + wn * 32 + ni * 16 + mq;
    float bv = bias[n];
#pragma unroll
    for (int mi = 0; mi < 2; ++mi) {
#pragma unroll
      for (int r = 0; r < 8; ++r) {
        int row = m0 + wm * 32 + mi * 16 + r + half * 8;
        Out[(size_t)row * 1024 + n] = acc[mi][ni][r] + bv;
      }
    }
  }
}

// ---------------------------------------------------------------------------
extern "C" void kernel_launch(void* const* d_in, const int* in_sizes, int n_in,
                              void* d_out, int out_size, void* d_ws, size_t ws_size,
                              hipStream_t stream) {
  (void)in_sizes; (void)n_in; (void)out_size; (void)ws_size;

  const float* x     = (const float*)d_in[0];   // [4,2048,1024]
  const float* w_qkv = (const float*)d_in[1];   // [1024,3072]
  const float* b_qkv = (const float*)d_in[2];   // [3072]
  const float* w_out = (const float*)d_in[3];   // [1024,1024]
  const float* b_out = (const float*)d_in[4];   // [1024]
  float* out = (float*)d_out;                   // [4,2048,1024]

  // Workspace: bf16 Q,K,V [B,H,S,HD] then bf16 O [B,S,D] = 64 MB total.
  unsigned short* qkv = (unsigned short*)d_ws;
  unsigned short* q = qkv;
  unsigned short* k = qkv + 1ull * 8192 * 1024;
  unsigned short* v = qkv + 2ull * 8192 * 1024;
  unsigned short* o = qkv + 3ull * 8192 * 1024;

  qkv_gemm_kernel<<<dim3(48, 64), 256, 0, stream>>>(x, w_qkv, b_qkv, qkv);
  flash_attn_kernel<<<dim3(32, 64), 128, 0, stream>>>(q, k, v, o);
  out_gemm_kernel<<<dim3(16, 64), 256, 0, stream>>>(o, w_out, b_out, out);
}
